// CausalSelfAttention_28114855919932
// MI455X (gfx1250) — matrix operand order repaired and checked by reading, not compiled, not run on hardware
//
#include <hip/hip_runtime.h>
#include <hip/hip_bf16.h>
#include <math.h>
#include <stdint.h>

#define B_   2
#define T_   2048
#define C_   1024
#define H_   16
#define HD_  64
#define M_   (B_ * T_)     // 4096 rows
#define NQKV (3 * C_)      // 3072

typedef __attribute__((ext_vector_type(16))) __bf16 v16bf;
typedef __attribute__((ext_vector_type(8)))  float  v8f;
typedef __attribute__((ext_vector_type(4)))  unsigned int v4u;
typedef __attribute__((ext_vector_type(8)))  int v8i;
typedef __attribute__((ext_vector_type(4)))  int v4i;

union Frag { v16bf v; unsigned u[8]; };

__device__ __forceinline__ unsigned short f2bf(float f) {
  unsigned u = __float_as_uint(f);
  u += 0x7FFFu + ((u >> 16) & 1u);           // round-to-nearest-even
  return (unsigned short)(u >> 16);
}

__device__ __forceinline__ v8f wmma_bf16(v16bf a, v16bf b, v8f c) {
#if defined(__gfx1250__)
  return __builtin_amdgcn_wmma_f32_16x16x32_bf16(
      false, a, false, b, (short)0, c, false, false);
#else
  return c;
#endif
}

// lane-xor reductions within 16-lane halves via ds_swizzle (group-of-32 mode)
// imm = xor<<10 | or<<5 | and ; and=0x1f keeps full sharing, xor in {1,2,4,8}
#define SWZ(x, imm) __int_as_float(__builtin_amdgcn_ds_swizzle(__float_as_int(x), imm))
__device__ __forceinline__ float red_max16(float x) {
  x = fmaxf(x, SWZ(x, 0x041f));
  x = fmaxf(x, SWZ(x, 0x081f));
  x = fmaxf(x, SWZ(x, 0x101f));
  x = fmaxf(x, SWZ(x, 0x201f));
  return x;
}
__device__ __forceinline__ float red_sum16(float x) {
  x += SWZ(x, 0x041f);
  x += SWZ(x, 0x081f);
  x += SWZ(x, 0x101f);
  x += SWZ(x, 0x201f);
  return x;
}

template<int N>
__device__ __forceinline__ void wait_tensor() {
#if defined(__gfx1250__)
  __builtin_amdgcn_s_wait_tensorcnt(N);     // ICE via template parameter
#endif
}

// TDM 2D strided-tile load to LDS. Dims/strides in elements (data_size = 2B).
// pad_amt >= 0 enables LDS padding: pad_int code (16 DW interval = 3, 32 DW = 4),
// pad_amt code (4 DW = 3) -> bank-conflict-free LDS row strides.
__device__ __forceinline__ void tdm_load_2d(unsigned lds_addr, const void* gptr,
                                            unsigned tile0, unsigned tile1,
                                            unsigned dim0, unsigned dim1,
                                            unsigned long long stride0,
                                            int pad_int, int pad_amt) {
#if defined(__gfx1250__)
  unsigned long long ga = (unsigned long long)gptr;
  v4u g0;
  g0.x = 1u;                                         // count=1, user descriptor
  g0.y = lds_addr;                                   // LDS byte address
  g0.z = (unsigned)ga;                               // global_addr[31:0]
  g0.w = (unsigned)((ga >> 32) & 0x1FFFFFFu) | (2u << 30);  // addr[56:32], type=2
  v8i g1;
  unsigned w0 = (1u << 16);                          // data_size = 2 bytes
  if (pad_amt >= 0)
    w0 |= (1u << 20) | ((unsigned)pad_int << 22) | ((unsigned)pad_amt << 25);
  g1[0] = (int)w0;
  g1[1] = (int)((dim0 & 0xFFFFu) << 16);             // tensor_dim0[15:0]
  g1[2] = (int)(((dim0 >> 16) & 0xFFFFu) | ((dim1 & 0xFFFFu) << 16));
  g1[3] = (int)(((dim1 >> 16) & 0xFFFFu) | ((tile0 & 0xFFFFu) << 16));
  g1[4] = (int)(tile1 & 0xFFFFu);                    // tile_dim1, tile_dim2=0
  g1[5] = (int)(unsigned)(stride0 & 0xFFFFFFFFull);  // tensor_dim0_stride[31:0]
  g1[6] = (int)(unsigned)((stride0 >> 32) & 0xFFFFull);
  g1[7] = 0;
  v4i g2 = {0, 0, 0, 0};                             // 2D: groups 2/3 unused
  v4i g3 = {0, 0, 0, 0};
  v8i g4 = {0, 0, 0, 0, 0, 0, 0, 0};
  __builtin_amdgcn_tensor_load_to_lds(g0, g1, g2, g3, g4, 0);
#endif
}

// ---------------- fp32 -> bf16 staging ----------------
__global__ __launch_bounds__(256)
void cvt4(const float4* __restrict__ s, ushort4* __restrict__ d, int n4) {
  int i = blockIdx.x * blockDim.x + threadIdx.x;
  if (i < n4) {
    float4 f = s[i];
    ushort4 o;
    o.x = f2bf(f.x); o.y = f2bf(f.y); o.z = f2bf(f.z); o.w = f2bf(f.w);
    d[i] = o;
  }
}

// fp32 [K][N] -> bf16 transposed [N][K] (coalesced writes along k)
__global__ __launch_bounds__(256)
void cvtT(const float* __restrict__ s, unsigned short* __restrict__ d, int K, int N) {
  int k = blockIdx.x * 64 + (threadIdx.x & 63);
  int n = blockIdx.y * 4 + (threadIdx.x >> 6);
  d[(size_t)n * K + k] = f2bf(s[(size_t)k * N + n]);
}

// ---------------- tiled WMMA GEMM, TDM-staged, double-buffered --------------
// A: bf16 [M][K] row-major.  Wt: bf16 [N][K] (W transposed).
// MODE 0: QKV gemm, scatter bf16 into Q/K/V [B,H,T,HD]
// MODE 1: proj gemm, fp32 out = A@W + bias
template<int MODE>
__global__ __launch_bounds__(256)
void gemm_bf16_kernel(const unsigned short* __restrict__ A,
                      const unsigned short* __restrict__ Wt,
                      const float* __restrict__ bias,
                      unsigned short* __restrict__ oQ,
                      unsigned short* __restrict__ oK,
                      unsigned short* __restrict__ oV,
                      float* __restrict__ oF,
                      int Kdim, int N) {
  const int LD = 40;                            // 32 data + 8 pad (bf16 units)
  __shared__ alignas(16) unsigned short ldsA[2][64 * 40];   // [m][k]
  __shared__ alignas(16) unsigned short ldsB[2][128 * 40];  // [n][k]

  const int tid  = threadIdx.x;
  const int lane = tid & 31, wv = tid >> 5;
  const int half = lane >> 4, l15 = lane & 15;
  const int wm = wv & 1, wn = wv >> 1;          // waves: 2 (rows) x 4 (cols)
  const int m0 = blockIdx.y * 64;
  const int n0 = blockIdx.x * 128;

  unsigned ldsA_off[2] = { (unsigned)(size_t)(void*)&ldsA[0][0],
                           (unsigned)(size_t)(void*)&ldsA[1][0] };
  unsigned ldsB_off[2] = { (unsigned)(size_t)(void*)&ldsB[0][0],
                           (unsigned)(size_t)(void*)&ldsB[1][0] };

  v8f acc[2][2];
#pragma unroll
  for (int i = 0; i < 2; i++)
#pragma unroll
    for (int j = 0; j < 2; j++)
#pragma unroll
      for (int e = 0; e < 8; e++) acc[i][j][e] = 0.f;

  const int nk = Kdim >> 5;
  if (wv == 0) {  // prologue: stage tile 0
    tdm_load_2d(ldsA_off[0], A + (size_t)m0 * Kdim, 32, 64,
                (unsigned)Kdim, (unsigned)M_, (unsigned long long)Kdim, 3, 3);
    tdm_load_2d(ldsB_off[0], Wt + (size_t)n0 * Kdim, 32, 128,
                (unsigned)Kdim, (unsigned)N, (unsigned long long)Kdim, 3, 3);
  }

  for (int it = 0; it < nk; it++) {
    const int buf = it & 1;
    if (wv == 0) {
      if (it + 1 < nk) {  // stage next tile into other buffer
        int k0 = (it + 1) << 5;
        tdm_load_2d(ldsA_off[buf ^ 1], A + (size_t)m0 * Kdim + k0, 32, 64,
                    (unsigned)Kdim, (unsigned)M_, (unsigned long long)Kdim, 3, 3);
        tdm_load_2d(ldsB_off[buf ^ 1], Wt + (size_t)n0 * Kdim + k0, 32, 128,
                    (unsigned)Kdim, (unsigned)N, (unsigned long long)Kdim, 3, 3);
        wait_tensor<2>();   // in-order: current tile's 2 ops have completed
      } else {
        wait_tensor<0>();
      }
    }
    __syncthreads();      // publish current tile to all waves

    Frag a[2], b[2];
#pragma unroll
    for (int ti = 0; ti < 2; ti++) {
      int r = wm * 32 + ti * 16 + l15;
#pragma unroll
      for (int i = 0; i < 8; i++) {
        int kk = half * 16 + 2 * i;
        a[ti].u[i] = *(const unsigned*)(&ldsA[buf][r * LD + kk]);
      }
    }
#pragma unroll
    for (int tj = 0; tj < 2; tj++) {
      int cN = wn * 32 + tj * 16 + l15;
#pragma unroll
      for (int i = 0; i < 8; i++)
        b[tj].u[i] = *(const unsigned*)(&ldsB[buf][cN * LD + half * 16 + 2 * i]);
    }
#pragma unroll
    for (int ti = 0; ti < 2; ti++)
#pragma unroll
      for (int tj = 0; tj < 2; tj++)
        acc[ti][tj] = wmma_bf16(a[ti].v, b[tj].v, acc[ti][tj]);

    __syncthreads();      // all reads done before buffer is overwritten
  }

#pragma unroll
  for (int ti = 0; ti < 2; ti++)
#pragma unroll
    for (int tj = 0; tj < 2; tj++)
#pragma unroll
      for (int r = 0; r < 8; r++) {
        int row = m0 + wm * 32 + ti * 16 + half * 8 + r;
        int col = n0 + wn * 32 + tj * 16 + l15;
        float v = acc[ti][tj][r] + bias[col];
        if (MODE == 0) {
          int which = col >> 10;          // 0=Q 1=K 2=V
          int cc = col & 1023;
          int hh = cc >> 6, d = cc & 63;
          int bb = row >> 11, t = row & 2047;
          unsigned short* dst = (which == 0) ? oQ : ((which == 1) ? oK : oV);
          dst[(((size_t)(bb * H_ + hh)) * T_ + t) * HD_ + d] = f2bf(v);
        } else {
          oF[(size_t)row * N + col] = v;
        }
      }
}

// ---------------- flash attention (causal, online softmax) ----------------
__global__ __launch_bounds__(128)
void attn_kernel(const unsigned short* __restrict__ Q,
                 const unsigned short* __restrict__ K,
                 const unsigned short* __restrict__ V,
                 unsigned short* __restrict__ Y) {
  __shared__ alignas(16) unsigned short ldsK[32 * 72];    // [kpos][hd], TDM pad
  __shared__ alignas(16) unsigned short ldsVt[64 * 40];   // [hd][kpos]
  __shared__ alignas(16) unsigned short ldsP[4][16 * 40]; // per wave [qrow][kpos]

  const int tid  = threadIdx.x;
  const int lane = tid & 31, wv = tid >> 5;          // 4 waves, 16 q-rows each
  const int half = lane >> 4, l15 = lane & 15;
  const int q0 = blockIdx.x * 64;
  const int h  = blockIdx.y, b = blockIdx.z;
  const size_t base = ((size_t)(b * H_ + h)) * T_ * HD_;
  const unsigned short* Qb = Q + base;
  const unsigned short* Kb = K + base;
  const unsigned short* Vb = V + base;
  const unsigned ldsK_off = (unsigned)(size_t)(void*)&ldsK[0];

  // Q A-fragments (loop invariant): frag f covers hd [32f, 32f+32)
  Frag aq[2];
  {
    int qrow = q0 + wv * 16 + l15;
#pragma unroll
    for (int f = 0; f < 2; f++)
#pragma unroll
      for (int i = 0; i < 8; i++) {
        int hd = f * 32 + half * 16 + 2 * i;
        aq[f].u[i] = *(const unsigned*)(Qb + (size_t)qrow * HD_ + hd);
      }
  }

  v8f o[4];                                 // O accum: hd cols 0-15 .. 48-63
#pragma unroll
  for (int c = 0; c < 4; c++)
#pragma unroll
    for (int e = 0; e < 8; e++) o[c][e] = 0.f;
  float mst[8], lst[8];
#pragma unroll
  for (int r = 0; r < 8; r++) { mst[r] = -INFINITY; lst[r] = 0.f; }

  const float scale = 0.125f;               // 1/sqrt(64)
  const int kend = q0 + 64;
  for (int kc = 0; kc < kend; kc += 32) {
    __syncthreads();
    // K tile via TDM (pad 32DW interval -> LDS row stride 36 DW = 72 bf16)
    if (wv == 0)
      tdm_load_2d(ldsK_off, Kb + (size_t)kc * HD_, 64, 32,
                  64u, (unsigned)T_, 64ull, 4, 3);
    // V tile transposed manually: [kpos][hd] -> [hd][kpos] (stride 40)
#pragma unroll
    for (int rep = 0; rep < 2; rep++) {
      int seg = tid + rep * 128;            // 256 segments of 8 bf16
      int row = seg >> 3, hs = (seg & 7) * 8;
      uint4 vvv = *(const uint4*)(Vb + (size_t)(kc + row) * HD_ + hs);
      const unsigned short* p = (const unsigned short*)&vvv;
#pragma unroll
      for (int j = 0; j < 8; j++) ldsVt[(hs + j) * 40 + row] = p[j];
    }
    if (wv == 0) wait_tensor<0>();
    __syncthreads();

    // S = Q K^T : two 16x16 tiles over kpos, HD contraction in two frags
    v8f s[2];
#pragma unroll
    for (int t = 0; t < 2; t++)
#pragma unroll
      for (int e = 0; e < 8; e++) s[t][e] = 0.f;
#pragma unroll
    for (int t = 0; t < 2; t++) {
      int kposl = t * 16 + l15;
#pragma unroll
      for (int f = 0; f < 2; f++) {
        Frag bk;
#pragma unroll
        for (int i = 0; i < 8; i++)
          bk.u[i] = *(const unsigned*)(&ldsK[kposl * 72 + f * 32 + half * 16 + 2 * i]);
        s[t] = wmma_bf16(aq[f].v, bk.v, s[t]);
      }
    }

    // online softmax per accumulator row r (row = q0+16wv+8*half+r)
#pragma unroll
    for (int r = 0; r < 8; r++) {
      int q = q0 + wv * 16 + half * 8 + r;
      float v0 = s[0][r] * scale, v1 = s[1][r] * scale;
      if (kc + l15 > q)      v0 = -INFINITY;
      if (kc + 16 + l15 > q) v1 = -INFINITY;
      float mx = red_max16(fmaxf(v0, v1));
      float mnew = fmaxf(mst[r], mx);
      float mn   = fmaxf(mnew, -1e30f);
      float corr = __expf(fmaxf(mst[r], -1e30f) - mn);
      float p0 = __expf(v0 - mn);
      float p1 = __expf(v1 - mn);
      float rs = red_sum16(p0 + p1);
      lst[r] = lst[r] * corr + rs;
      mst[r] = mnew;
#pragma unroll
      for (int c = 0; c < 4; c++) o[c][r] *= corr;
      int prow = half * 8 + r;              // C-layout -> LDS [qrow][kpos]
      ldsP[wv][prow * 40 + l15]      = f2bf(p0);
      ldsP[wv][prow * 40 + 16 + l15] = f2bf(p1);
    }
    __syncthreads();

    // O += P @ V : one A-frag (16x32), 4 B-frags over hd columns
    Frag ap;
#pragma unroll
    for (int i = 0; i < 8; i++) {
      int kk = half * 16 + 2 * i;
      ap.u[i] = *(const unsigned*)(&ldsP[wv][l15 * 40 + kk]);
    }
#pragma unroll
    for (int c = 0; c < 4; c++) {
      Frag bv;
#pragma unroll
      for (int i = 0; i < 8; i++)
        bv.u[i] = *(const unsigned*)(&ldsVt[(c * 16 + l15) * 40 + half * 16 + 2 * i]);
      o[c] = wmma_bf16(ap.v, bv.v, o[c]);
    }
  }

  // normalize, write y in [B,T,C] layout (bf16 feeds proj GEMM)
#pragma unroll
  for (int r = 0; r < 8; r++) {
    float inv = 1.0f / lst[r];
    int q = q0 + wv * 16 + half * 8 + r;
    size_t rowo = ((size_t)b * T_ + q) * C_ + (size_t)h * HD_;
#pragma unroll
    for (int c = 0; c < 4; c++)
      Y[rowo + c * 16 + l15] = f2bf(o[c][r] * inv);
  }
}

extern "C" void kernel_launch(void* const* d_in, const int* in_sizes, int n_in,
                              void* d_out, int out_size, void* d_ws, size_t ws_size,
                              hipStream_t stream) {
  const float* x      = (const float*)d_in[0];
  const float* w_attn = (const float*)d_in[1];
  const float* b_attn = (const float*)d_in[2];
  const float* w_proj = (const float*)d_in[3];
  const float* b_proj = (const float*)d_in[4];
  float* out = (float*)d_out;

  unsigned short* ws = (unsigned short*)d_ws;
  size_t off = 0;
  unsigned short* xb  = ws + off; off += (size_t)M_ * C_;      // x bf16
  unsigned short* wab = ws + off; off += (size_t)C_ * NQKV;    // w_attn^T bf16
  unsigned short* wpb = ws + off; off += (size_t)C_ * C_;      // w_proj^T bf16
  unsigned short* Qb  = ws + off; off += (size_t)M_ * C_;      // Q [B,H,T,HD]
  unsigned short* Kb  = ws + off; off += (size_t)M_ * C_;      // K [B,H,T,HD]
  unsigned short* Vb  = ws + off; off += (size_t)M_ * C_;      // V [B,H,T,HD]
  unsigned short* yb  = xb;  // alias: xb dead after QKV GEMM

  cvt4<<<(M_ * C_ / 4 + 255) / 256, 256, 0, stream>>>(
      (const float4*)x, (ushort4*)xb, M_ * C_ / 4);
  cvtT<<<dim3(C_ / 64, NQKV / 4), 256, 0, stream>>>(w_attn, wab, C_, NQKV);
  cvtT<<<dim3(C_ / 64, C_ / 4), 256, 0, stream>>>(w_proj, wpb, C_, C_);

  gemm_bf16_kernel<0><<<dim3(NQKV / 128, M_ / 64), 256, 0, stream>>>(
      xb, wab, b_attn, Qb, Kb, Vb, nullptr, C_, NQKV);

  attn_kernel<<<dim3(T_ / 64, H_, B_), 128, 0, stream>>>(Qb, Kb, Vb, yb);

  gemm_bf16_kernel<1><<<dim3(C_ / 128, M_ / 64), 256, 0, stream>>>(
      yb, wpb, b_proj, nullptr, nullptr, nullptr, out, C_, C_);
}
